// Mamba2Layer_37580963840666
// MI455X (gfx1250) — compile-verified, matrix operand order checked by reading, not run on hardware
//
#include <hip/hip_runtime.h>

#define BATCH_  8
#define SEQ_    2048
#define DMODEL_ 1024
#define DSTATE_ 128
#define DINNER_ 2048
#define NHEADS_ 32
#define HEADDIM_ 64
#define CONVDIM_ 2304
#define DINPROJ_ 4384
#define NPAD_   4480          // 4384 padded to a multiple of 128
#define CHUNK_  64
#define NC_     32            // SEQ / CHUNK
#define ROWS_   16384         // BATCH * SEQ
#define EPS_    1e-5f

typedef __attribute__((ext_vector_type(16))) __bf16         bf16x16;
typedef __attribute__((ext_vector_type(8)))  float          f32x8;
typedef __attribute__((ext_vector_type(8)))  unsigned short u16x8;
typedef __attribute__((ext_vector_type(4)))  unsigned int   u32x4;
typedef __attribute__((ext_vector_type(8)))  int            i32x8;
typedef __attribute__((ext_vector_type(4)))  int            i32x4;

union Frag {
  bf16x16 v;
  u16x8   h[2];
};
union Acc {
  f32x8 v;
  float f[8];
};

__device__ __forceinline__ unsigned short f2bf(float f) {
  unsigned int u = __builtin_bit_cast(unsigned int, f);
  u += 0x7FFFu + ((u >> 16) & 1u);       // round-to-nearest-even
  return (unsigned short)(u >> 16);
}
__device__ __forceinline__ float bf2f(unsigned short h) {
  unsigned int u = ((unsigned int)h) << 16;
  return __builtin_bit_cast(float, u);
}
__device__ __forceinline__ float sigmoidf_(float x) { return 1.0f / (1.0f + __expf(-x)); }

// Load one 16x32 bf16 A/B fragment from a row-major [rows x stride] bf16 array.
// Per ISA 7.12.2: lane group hi (lane>=16) holds K = k0+hi*8 + {0..7} and K = k0+16+hi*8 + {0..7}.
// Caller passes k0 = kbase + hi*8; both chunks are contiguous 16-byte loads.
__device__ __forceinline__ void load_frag(Frag& f, const unsigned short* p, int stride, int row, int k0) {
  const unsigned short* q = p + row * stride + k0;
  f.h[0] = *(const u16x8*)(q);
  f.h[1] = *(const u16x8*)(q + 16);
}

__device__ __forceinline__ void wmma_bf16(Acc& acc, const Frag& a, const Frag& b) {
  acc.v = __builtin_amdgcn_wmma_f32_16x16x32_bf16(
      false, a.v, false, b.v, (short)0, acc.v, false, false);
}

// ---------------------------------------------------------------------------
// TDM: DMA one 128-row x 32-col bf16 tile (row stride = stride_elems) from a
// row-major global matrix into LDS (rows packed contiguously, 64B per row).
// D# per CDNA5 ISA: group0 = {count, lds_addr,
// global_addr, type=2}; group1 = {data_size=2B, tensor dims, tile dims,
// dim0 stride}; groups 2/3 zero (2D tile). Tracked by TENSORcnt.
// This toolchain uses the 6-arg builtin form:
//   (uint32x4 g0, int32x8 g1, int32x4 g2, int32x4 g3, int32x8 extra, i32 cpol)
// ---------------------------------------------------------------------------
__device__ __forceinline__ void tdm_load_tile_128x32(unsigned lds_off,
                                                     const unsigned short* g,
                                                     int stride_elems) {
  unsigned long long ga = (unsigned long long)g;
  u32x4 g0;
  g0[0] = 1u;                                        // count=1, user mode
  g0[1] = lds_off;                                   // LDS byte address
  g0[2] = (unsigned)ga;                              // global_addr[31:0]
  g0[3] = (unsigned)(ga >> 32) | 0x80000000u;        // global_addr[56:32] | type=2<<30

  const unsigned td0 = 32u, td1 = 128u;              // tensor dims (exact tile: no OOB)
  const unsigned tile0 = 32u, tile1 = 128u;          // tile dims (elements, rows)
  unsigned long long st0 = (unsigned long long)stride_elems;
  i32x8 g1;
  g1[0] = (int)(1u << 16);                                            // data_size=1 -> 2 bytes
  g1[1] = (int)((td0 & 0xFFFFu) << 16);                               // ab_addr=0 | td0.lo
  g1[2] = (int)(((td0 >> 16) & 0xFFFFu) | ((td1 & 0xFFFFu) << 16));   // td0.hi | td1.lo
  g1[3] = (int)(((td1 >> 16) & 0xFFFFu) | (tile0 << 16));             // td1.hi | tile_dim0
  g1[4] = (int)(tile1 & 0xFFFFu);                                     // tile_dim1 | tile_dim2=0
  g1[5] = (int)(unsigned)st0;                                         // dim0_stride[31:0]
  g1[6] = (int)((unsigned)(st0 >> 32) & 0xFFFFu);                     // dim0_stride[47:32]
  g1[7] = 0;                                                          // dim1_stride=0
  i32x4 gz4;
  gz4[0] = 0; gz4[1] = 0; gz4[2] = 0; gz4[3] = 0;
  i32x8 gz8;
  for (int i = 0; i < 8; ++i) gz8[i] = 0;
  __builtin_amdgcn_tensor_load_to_lds(g0, g1, gz4, gz4, gz8, 0);
}

// ---------------------------------------------------------------------------
// 1) LayerNorm (f32 in) -> bf16 out
// ---------------------------------------------------------------------------
__global__ __launch_bounds__(256) void layernorm_kernel(
    const float* __restrict__ x, const float* __restrict__ w, const float* __restrict__ b,
    unsigned short* __restrict__ out)
{
  __shared__ float red[256];
  const int row = blockIdx.x;
  const int tid = threadIdx.x;
  const float* xr = x + (size_t)row * DMODEL_;
  float v[4];
  float s = 0.f;
  for (int j = 0; j < 4; ++j) { v[j] = xr[tid + j * 256]; s += v[j]; }
  red[tid] = s;
  __syncthreads();
  for (int o = 128; o > 0; o >>= 1) { if (tid < o) red[tid] += red[tid + o]; __syncthreads(); }
  float mu = red[0] / (float)DMODEL_;
  __syncthreads();
  s = 0.f;
  for (int j = 0; j < 4; ++j) { float d = v[j] - mu; s += d * d; }
  red[tid] = s;
  __syncthreads();
  for (int o = 128; o > 0; o >>= 1) { if (tid < o) red[tid] += red[tid + o]; __syncthreads(); }
  float rs = rsqrtf(red[0] / (float)DMODEL_ + EPS_);
  for (int j = 0; j < 4; ++j) {
    int c = tid + j * 256;
    out[(size_t)row * DMODEL_ + c] = f2bf((v[j] - mu) * rs * w[c] + b[c]);
  }
}

// ---------------------------------------------------------------------------
// 2) f32 -> bf16 convert with zero row padding (for weights)
// ---------------------------------------------------------------------------
__global__ void convert_pad_kernel(const float* __restrict__ src, unsigned short* __restrict__ dst,
                                   int rows, int cols, int dst_rows)
{
  int idx = blockIdx.x * blockDim.x + threadIdx.x;
  int total = dst_rows * cols;
  if (idx >= total) return;
  int r = idx / cols;
  dst[idx] = (r < rows) ? f2bf(src[idx]) : (unsigned short)0;
}

// ---------------------------------------------------------------------------
// 3) bf16 WMMA GEMM:  C[M,N] = A[M,K] * B[N,K]^T  (+ optional f32 residual)
//    Block tile 128x128, 8 waves, each wave 64x32 (4x2 accumulator tiles).
//    Tiles are DMA'd into double-buffered LDS by the Tensor Data Mover
//    (wave 0 issues descriptors; TENSORcnt gates the buffer swap), so the
//    compute waves run a pure ds_load + v_wmma inner loop.
// ---------------------------------------------------------------------------
__global__ __launch_bounds__(256) void gemm_bf16_kernel(
    const unsigned short* __restrict__ A, const unsigned short* __restrict__ Bw,
    float* __restrict__ C, const float* __restrict__ resid,
    int M, int N, int K)
{
  __shared__ unsigned short As[2][128 * 32];
  __shared__ unsigned short Bs[2][128 * 32];

  const int tid  = threadIdx.x;
  const int bm   = blockIdx.y * 128;
  const int bn   = blockIdx.x * 128;
  const int lane = tid & 31;
  const int wave = tid >> 5;
  const int wr   = wave >> 2;          // 0..1
  const int wc   = wave & 3;           // 0..3
  const int mi   = lane & 15;
  const int hi   = lane >> 4;

  const unsigned ldsA[2] = { (unsigned)(unsigned long long)&As[0][0],
                             (unsigned)(unsigned long long)&As[1][0] };
  const unsigned ldsB[2] = { (unsigned)(unsigned long long)&Bs[0][0],
                             (unsigned)(unsigned long long)&Bs[1][0] };

  Acc acc[4][2];
  for (int i = 0; i < 4; ++i)
    for (int j = 0; j < 2; ++j)
      for (int r = 0; r < 8; ++r) acc[i][j].f[r] = 0.f;

  const int nsteps = K / 32;

  if (wave == 0) {
    tdm_load_tile_128x32(ldsA[0], A + (size_t)bm * K, K);
    tdm_load_tile_128x32(ldsB[0], Bw + (size_t)bn * K, K);
  }

  for (int step = 0; step < nsteps; ++step) {
    const int cur = step & 1;
    if (wave == 0) {
      if (step + 1 < nsteps) {
        // prefetch next k-slab into the other buffer, then wait for current
        tdm_load_tile_128x32(ldsA[cur ^ 1], A + (size_t)bm * K + (step + 1) * 32, K);
        tdm_load_tile_128x32(ldsB[cur ^ 1], Bw + (size_t)bn * K + (step + 1) * 32, K);
        __builtin_amdgcn_s_wait_tensorcnt(2);   // in-order: current buffer landed
      } else {
        __builtin_amdgcn_s_wait_tensorcnt(0);
      }
    }
    __syncthreads();

    Frag af[4], bf[2];
    for (int i = 0; i < 4; ++i) load_frag(af[i], &As[cur][0], 32, wr * 64 + i * 16 + mi, hi * 8);
    for (int j = 0; j < 2; ++j) load_frag(bf[j], &Bs[cur][0], 32, wc * 32 + j * 16 + mi, hi * 8);
    for (int i = 0; i < 4; ++i)
      for (int j = 0; j < 2; ++j)
        wmma_bf16(acc[i][j], af[i], bf[j]);
    __syncthreads();   // all waves done with 'cur' before TDM overwrites it
  }

  for (int i = 0; i < 4; ++i) {
    for (int j = 0; j < 2; ++j) {
      int col = bn + wc * 32 + j * 16 + mi;
      for (int r = 0; r < 8; ++r) {
        int row = bm + wr * 64 + i * 16 + r + hi * 8;
        float v = acc[i][j].f[r];
        if (resid) v += resid[(size_t)row * N + col];
        C[(size_t)row * N + col] = v;
      }
    }
  }
}

// ---------------------------------------------------------------------------
// 4) depthwise conv(4 taps) + SiLU + softplus(dt); emit bf16 x*dt, B, C; f32 xs
// ---------------------------------------------------------------------------
__global__ __launch_bounds__(256) void conv_silu_kernel(
    const float* __restrict__ zxbcdt, const float* __restrict__ conv_w,
    const float* __restrict__ conv_b, const float* __restrict__ dt_bias,
    float* __restrict__ dt_soft, float* __restrict__ xs,
    unsigned short* __restrict__ xdt, unsigned short* __restrict__ Bo,
    unsigned short* __restrict__ Co)
{
  __shared__ float dtsh[NHEADS_];
  const int row = blockIdx.x;
  const int b   = row / SEQ_;
  const int l   = row % SEQ_;
  const int tid = threadIdx.x;
  if (tid < NHEADS_) {
    float d = zxbcdt[(size_t)row * NPAD_ + (DINNER_ + CONVDIM_) + tid] + dt_bias[tid];
    float sp = (d > 20.f) ? d : log1pf(__expf(d));
    dtsh[tid] = sp;
    dt_soft[(size_t)row * NHEADS_ + tid] = sp;
  }
  __syncthreads();
  for (int ch = tid; ch < CONVDIM_; ch += 256) {
    float a = conv_b[ch];
#pragma unroll
    for (int k = 0; k < 4; ++k) {
      int ll = l - 3 + k;
      if (ll >= 0)
        a += zxbcdt[(size_t)(b * SEQ_ + ll) * NPAD_ + DINNER_ + ch] * conv_w[ch * 4 + k];
    }
    float v = a * sigmoidf_(a);
    if (ch < DINNER_) {
      int hh = ch >> 6;
      xs[(size_t)row * DINNER_ + ch]  = v;
      xdt[(size_t)row * DINNER_ + ch] = f2bf(v * dtsh[hh]);
    } else if (ch < DINNER_ + DSTATE_) {
      Bo[(size_t)row * DSTATE_ + (ch - DINNER_)] = f2bf(v);
    } else {
      Co[(size_t)row * DSTATE_ + (ch - DINNER_ - DSTATE_)] = f2bf(v);
    }
  }
}

// ---------------------------------------------------------------------------
// 5) SSD-A per (b, chunk): G = C*B^T (once), then per head:
//    Y_diag = (G .* L) * X   and   state = X^T * (decay .* B)
// ---------------------------------------------------------------------------
__global__ __launch_bounds__(256) void ssd_a_kernel(
    const unsigned short* __restrict__ Bg, const unsigned short* __restrict__ Cg,
    const unsigned short* __restrict__ xdt, const float* __restrict__ dt_soft,
    const float* __restrict__ A_log,
    float* __restrict__ acs_g, float* __restrict__ states, float* __restrict__ Y)
{
  __shared__ unsigned short Bsh[64 * 128];   // B[l][n]
  __shared__ float          Gs[64 * 64];     // C*B^T (f32)
  __shared__ unsigned short Ms[64 * 64];     // (G.*L)[l][s] bf16
  __shared__ unsigned short XT[64 * 64];     // X^T[p][l] bf16
  __shared__ unsigned short BdT[128 * 64];   // (decay.*B)^T[n][l] bf16
  __shared__ float acs_sh[64];
  __shared__ float decay_sh[64];

  const int tid  = threadIdx.x;
  const int b    = blockIdx.x / NC_;
  const int c    = blockIdx.x % NC_;
  const int row0 = b * SEQ_ + c * CHUNK_;
  const int lane = tid & 31, wave = tid >> 5;
  const int mi   = lane & 15, hi = lane >> 4;

  for (int i = tid; i < (64 * 128) / 8; i += 256)
    ((u16x8*)Bsh)[i] = ((const u16x8*)(Bg + (size_t)row0 * DSTATE_))[i];
  __syncthreads();

  // G = C * B^T : 16 tiles of 16x16, K = 128
  for (int t = wave * 2; t < wave * 2 + 2; ++t) {
    const int tm = (t >> 2) * 16, tn = (t & 3) * 16;
    Acc g;
    for (int r = 0; r < 8; ++r) g.f[r] = 0.f;
    for (int kb = 0; kb < DSTATE_; kb += 32) {
      Frag a, bb;
      load_frag(a,  Cg + (size_t)row0 * DSTATE_, DSTATE_, tm + mi, kb + hi * 8);
      load_frag(bb, Bsh,                         DSTATE_, tn + mi, kb + hi * 8);
      wmma_bf16(g, a, bb);
    }
    for (int r = 0; r < 8; ++r)
      Gs[(tm + r + hi * 8) * 64 + tn + mi] = g.f[r];
  }
  __syncthreads();

  for (int h = 0; h < NHEADS_; ++h) {
    if (tid < 64) {
      float a = -__expf(A_log[h]);
      acs_sh[tid] = a * dt_soft[(size_t)(row0 + tid) * NHEADS_ + h];
    }
    __syncthreads();
    if (tid == 0) {
      float run = 0.f;
      for (int i = 0; i < 64; ++i) { run += acs_sh[i]; acs_sh[i] = run; }
    }
    __syncthreads();
    if (tid < 64) {
      acs_g[(size_t)(((b * NC_ + c) * NHEADS_) + h) * CHUNK_ + tid] = acs_sh[tid];
      decay_sh[tid] = __expf(acs_sh[63] - acs_sh[tid]);
    }
    __syncthreads();

    for (int idx = tid; idx < 64 * 64; idx += 256) {
      int l = idx >> 6, s = idx & 63;
      float lm = (s <= l) ? __expf(acs_sh[l] - acs_sh[s]) : 0.f;
      Ms[idx] = f2bf(Gs[idx] * lm);
    }
    for (int idx = tid; idx < 64 * 64; idx += 256) {
      int p = idx >> 6, l = idx & 63;
      XT[idx] = xdt[(size_t)(row0 + l) * DINNER_ + h * HEADDIM_ + p];
    }
    for (int idx = tid; idx < 128 * 64; idx += 256) {
      int n = idx >> 6, l = idx & 63;
      BdT[idx] = f2bf(decay_sh[l] * bf2f(Bsh[l * DSTATE_ + n]));
    }
    __syncthreads();

    // Y_diag = M * X : 16 tiles (l x p), K = 64
    for (int t = wave * 2; t < wave * 2 + 2; ++t) {
      const int tm = (t >> 2) * 16, tn = (t & 3) * 16;
      Acc y;
      for (int r = 0; r < 8; ++r) y.f[r] = 0.f;
      for (int kb = 0; kb < CHUNK_; kb += 32) {
        Frag a, bb;
        load_frag(a,  Ms, 64, tm + mi, kb + hi * 8);
        load_frag(bb, XT, 64, tn + mi, kb + hi * 8);
        wmma_bf16(y, a, bb);
      }
      for (int r = 0; r < 8; ++r) {
        int l = tm + r + hi * 8;
        Y[(size_t)(row0 + l) * DINNER_ + h * HEADDIM_ + tn + mi] = y.f[r];
      }
    }

    // state = X^T * (decay .* B) : 64x128 = 32 tiles, K = 64
    float* st = states + (size_t)(((b * NC_ + c) * NHEADS_) + h) * (HEADDIM_ * DSTATE_);
    for (int t = wave * 4; t < wave * 4 + 4; ++t) {
      const int tp = (t >> 3) * 16, tn = (t & 7) * 16;
      Acc sa;
      for (int r = 0; r < 8; ++r) sa.f[r] = 0.f;
      for (int kb = 0; kb < CHUNK_; kb += 32) {
        Frag a, bb;
        load_frag(a,  XT,  64, tp + mi, kb + hi * 8);
        load_frag(bb, BdT, 64, tn + mi, kb + hi * 8);
        wmma_bf16(sa, a, bb);
      }
      for (int r = 0; r < 8; ++r) {
        int p = tp + r + hi * 8;
        st[p * DSTATE_ + tn + mi] = sa.f[r];
      }
    }
    __syncthreads();
  }
}

// ---------------------------------------------------------------------------
// 6) inter-chunk state recurrence (in place: states[c] -> states_prev[c])
// ---------------------------------------------------------------------------
__global__ __launch_bounds__(256) void rec_kernel(float* __restrict__ states,
                                                  const float* __restrict__ acs_g)
{
  const int tid = threadIdx.x;
  const int b = blockIdx.x / NHEADS_;
  const int h = blockIdx.x % NHEADS_;
  float S[32];
  for (int j = 0; j < 32; ++j) S[j] = 0.f;
  for (int c = 0; c < NC_; ++c) {
    float* st = states + (size_t)(((b * NC_ + c) * NHEADS_) + h) * (HEADDIM_ * DSTATE_);
    float d = __expf(acs_g[(size_t)(((b * NC_ + c) * NHEADS_) + h) * CHUNK_ + 63]);
    for (int j = 0; j < 32; ++j) {
      int idx = tid + j * 256;
      float cs = st[idx];
      st[idx] = S[j];               // states_prev for chunk c
      S[j] = d * S[j] + cs;
    }
  }
}

// ---------------------------------------------------------------------------
// 7) SSD-B per (b, chunk): Y += (C * S_prev^T) .* exp(A_cs)
// ---------------------------------------------------------------------------
__global__ __launch_bounds__(256) void ssd_b_kernel(
    const unsigned short* __restrict__ Cg, const float* __restrict__ states,
    const float* __restrict__ acs_g, float* __restrict__ Y)
{
  __shared__ unsigned short Ss[64 * 128];   // S_prev[p][n] bf16
  __shared__ float sdo_sh[64];
  const int tid  = threadIdx.x;
  const int b    = blockIdx.x / NC_;
  const int c    = blockIdx.x % NC_;
  const int row0 = b * SEQ_ + c * CHUNK_;
  const int lane = tid & 31, wave = tid >> 5;
  const int mi   = lane & 15, hi = lane >> 4;

  for (int h = 0; h < NHEADS_; ++h) {
    const float* st = states + (size_t)(((b * NC_ + c) * NHEADS_) + h) * (HEADDIM_ * DSTATE_);
    if (tid < 64)
      sdo_sh[tid] = __expf(acs_g[(size_t)(((b * NC_ + c) * NHEADS_) + h) * CHUNK_ + tid]);
    for (int idx = tid; idx < 64 * 128; idx += 256)
      Ss[idx] = f2bf(st[idx]);
    __syncthreads();

    for (int t = wave * 2; t < wave * 2 + 2; ++t) {
      const int tm = (t >> 2) * 16, tn = (t & 3) * 16;
      Acc y;
      for (int r = 0; r < 8; ++r) y.f[r] = 0.f;
      for (int kb = 0; kb < DSTATE_; kb += 32) {
        Frag a, bb;
        load_frag(a,  Cg + (size_t)row0 * DSTATE_, DSTATE_, tm + mi, kb + hi * 8);
        load_frag(bb, Ss,                          DSTATE_, tn + mi, kb + hi * 8);
        wmma_bf16(y, a, bb);
      }
      for (int r = 0; r < 8; ++r) {
        int l = tm + r + hi * 8;
        float* yp = &Y[(size_t)(row0 + l) * DINNER_ + h * HEADDIM_ + tn + mi];
        *yp += y.f[r] * sdo_sh[l];
      }
    }
    __syncthreads();
  }
}

// ---------------------------------------------------------------------------
// 8) Y += xs*D, gate with silu(z), RMSNorm, emit bf16
// ---------------------------------------------------------------------------
__global__ __launch_bounds__(256) void gate_rms_kernel(
    const float* __restrict__ Y, const float* __restrict__ xs,
    const float* __restrict__ Dp, const float* __restrict__ zxbcdt,
    const float* __restrict__ rms_w, unsigned short* __restrict__ yout)
{
  __shared__ float red[256];
  const int row = blockIdx.x;
  const int tid = threadIdx.x;
  float g[8];
  float s = 0.f;
  for (int j = 0; j < 8; ++j) {
    int ch = tid + j * 256;
    float v = Y[(size_t)row * DINNER_ + ch] + xs[(size_t)row * DINNER_ + ch] * Dp[ch >> 6];
    float z = zxbcdt[(size_t)row * NPAD_ + ch];
    float gg = v * (z * sigmoidf_(z));
    g[j] = gg;
    s += gg * gg;
  }
  red[tid] = s;
  __syncthreads();
  for (int o = 128; o > 0; o >>= 1) { if (tid < o) red[tid] += red[tid + o]; __syncthreads(); }
  float rs = rsqrtf(red[0] / (float)DINNER_ + EPS_);
  for (int j = 0; j < 8; ++j) {
    int ch = tid + j * 256;
    yout[(size_t)row * DINNER_ + ch] = f2bf(g[j] * rs * rms_w[ch]);
  }
}

// ---------------------------------------------------------------------------
extern "C" void kernel_launch(void* const* d_in, const int* in_sizes, int n_in,
                              void* d_out, int out_size, void* d_ws, size_t ws_size,
                              hipStream_t stream)
{
  const float* x       = (const float*)d_in[0];
  const float* ln_w    = (const float*)d_in[1];
  const float* ln_b    = (const float*)d_in[2];
  const float* W_in    = (const float*)d_in[3];
  const float* conv_w  = (const float*)d_in[4];
  const float* conv_b  = (const float*)d_in[5];
  const float* dt_bias = (const float*)d_in[6];
  const float* A_log   = (const float*)d_in[7];
  const float* Dp      = (const float*)d_in[8];
  const float* rms_w   = (const float*)d_in[9];
  const float* W_out   = (const float*)d_in[10];
  float* out = (float*)d_out;

  char* ws = (char*)d_ws;
  size_t off = 0;
  auto alloc = [&](size_t bytes) -> void* {
    void* p = (void*)(ws + off);
    off += (bytes + 255) & ~(size_t)255;
    return p;
  };

  unsigned short* xn_bf   = (unsigned short*)alloc((size_t)ROWS_ * DMODEL_ * 2);
  unsigned short* win_bf  = (unsigned short*)alloc((size_t)NPAD_ * DMODEL_ * 2);
  unsigned short* wout_bf = (unsigned short*)alloc((size_t)DMODEL_ * DINNER_ * 2);
  float* zxbcdt           = (float*)alloc((size_t)ROWS_ * NPAD_ * 4);
  float* dt_soft          = (float*)alloc((size_t)ROWS_ * NHEADS_ * 4);
  float* xs               = (float*)alloc((size_t)ROWS_ * DINNER_ * 4);
  unsigned short* xdt     = (unsigned short*)alloc((size_t)ROWS_ * DINNER_ * 2);
  unsigned short* Bo      = (unsigned short*)alloc((size_t)ROWS_ * DSTATE_ * 2);
  unsigned short* Co      = (unsigned short*)alloc((size_t)ROWS_ * DSTATE_ * 2);
  float* acs              = (float*)alloc((size_t)BATCH_ * NC_ * NHEADS_ * CHUNK_ * 4);
  float* states           = (float*)alloc((size_t)BATCH_ * NC_ * NHEADS_ * HEADDIM_ * DSTATE_ * 4);
  float* Ybuf             = (float*)alloc((size_t)ROWS_ * DINNER_ * 4);
  unsigned short* ygate   = (unsigned short*)alloc((size_t)ROWS_ * DINNER_ * 2);

  layernorm_kernel<<<ROWS_, 256, 0, stream>>>(x, ln_w, ln_b, xn_bf);

  {
    int total = NPAD_ * DMODEL_;
    convert_pad_kernel<<<(total + 255) / 256, 256, 0, stream>>>(W_in, win_bf, DINPROJ_, DMODEL_, NPAD_);
  }
  {
    int total = DMODEL_ * DINNER_;
    convert_pad_kernel<<<(total + 255) / 256, 256, 0, stream>>>(W_out, wout_bf, DMODEL_, DINNER_, DMODEL_);
  }

  // in-proj: [16384 x 1024] * [4480 x 1024]^T -> [16384 x 4480]
  gemm_bf16_kernel<<<dim3(NPAD_ / 128, ROWS_ / 128), 256, 0, stream>>>(
      xn_bf, win_bf, zxbcdt, nullptr, ROWS_, NPAD_, DMODEL_);

  conv_silu_kernel<<<ROWS_, 256, 0, stream>>>(zxbcdt, conv_w, conv_b, dt_bias,
                                              dt_soft, xs, xdt, Bo, Co);

  ssd_a_kernel<<<BATCH_ * NC_, 256, 0, stream>>>(Bo, Co, xdt, dt_soft, A_log, acs, states, Ybuf);

  rec_kernel<<<BATCH_ * NHEADS_, 256, 0, stream>>>(states, acs);

  ssd_b_kernel<<<BATCH_ * NC_, 256, 0, stream>>>(Co, states, acs, Ybuf);

  gate_rms_kernel<<<ROWS_, 256, 0, stream>>>(Ybuf, xs, Dp, zxbcdt, rms_w, ygate);

  // out-proj with fused residual: [16384 x 2048] * [1024 x 2048]^T + x
  gemm_bf16_kernel<<<dim3(DMODEL_ / 128, ROWS_ / 128), 256, 0, stream>>>(
      ygate, wout_bf, out, x, ROWS_, DMODEL_, DINNER_);

  (void)in_sizes; (void)n_in; (void)out_size; (void)ws_size;
}
